// Attn_2869038154087
// MI455X (gfx1250) — matrix operand order repaired and checked by reading, not compiled, not run on hardware
//
#include <hip/hip_runtime.h>
#include <hip/hip_bf16.h>

#define B_  2
#define S_  2048
#define D_  1024
#define H_  16
#define DK_ 64

typedef __attribute__((ext_vector_type(16))) __bf16 v16bf;
typedef __attribute__((ext_vector_type(8)))  float  v8f;
typedef __attribute__((__vector_size__(4 * sizeof(int)))) int v4i;

union FragAB { v16bf v; uint4 q[2]; };

#if __has_builtin(__builtin_amdgcn_global_load_async_to_lds_b128) && \
    __has_builtin(__builtin_amdgcn_s_wait_asynccnt)
#define HAS_ASYNC_LDS 1
#else
#define HAS_ASYNC_LDS 0
#endif

__device__ __forceinline__ unsigned short f2bf(float f) {
  unsigned int u = __float_as_uint(f);
  u += 0x7fffu + ((u >> 16) & 1u);   // round to nearest even
  return (unsigned short)(u >> 16);
}

// 16-byte global -> LDS copy; async (ASYNCcnt-tracked) when available.
__device__ __forceinline__ void copy16_g2l(const unsigned short* g,
                                           unsigned short* l) {
#if HAS_ASYNC_LDS
  __builtin_amdgcn_global_load_async_to_lds_b128(
      (__attribute__((address_space(1))) v4i*)(v4i*)(void*)
          const_cast<unsigned short*>(g),
      (__attribute__((address_space(3))) v4i*)(v4i*)(void*)l,
      0, 0);
#else
  *(uint4*)l = *(const uint4*)g;
#endif
}

__device__ __forceinline__ void async_wait0() {
#if HAS_ASYNC_LDS
  __builtin_amdgcn_s_wait_asynccnt(0);
#endif
}

__device__ __forceinline__ v8f wmma_bf16(const FragAB& a, const FragAB& b, v8f c) {
  return __builtin_amdgcn_wmma_f32_16x16x32_bf16(false, a.v, false, b.v,
                                                 (short)0, c, false, false);
}

// ---------------------------------------------------------------------------
// GEMM: out[M,N] = (A[M,K] @ W[K,N] + bias[N]) * scale
// A is fp32 (TIn=float) or bf16-in-ushort (TIn=unsigned short).
// Block tile 128x128, 8 waves in 4x2, each computing 32x64 (2x4 WMMA tiles)
// -> 8 WMMAs per K-step per wave. LDS: A row-major [128][32] bf16, W staged
// transposed [128][32] bf16 so all fragment reads are contiguous b128.
// ---------------------------------------------------------------------------
template <typename TIn, bool OUT_BF16>
__global__ __launch_bounds__(256) void gemm_wmma_kernel(
    const TIn* __restrict__ A, const float* __restrict__ W,
    const float* __restrict__ bias, void* __restrict__ outp,
    int M, int N, int K, float scale) {
  constexpr int BM = 128, BN = 128, BK = 32;
  __shared__ unsigned short As[BM][BK];   // 8 KB
  __shared__ unsigned short Bt[BN][BK];   // 8 KB (transposed W tile)

  const int tid  = threadIdx.x;
  const int wave = tid >> 5, lane = tid & 31;
  const int ln = lane & 15, kh = lane >> 4;
  const int wm = wave >> 1, wn = wave & 1;          // 4 x 2 wave grid
  const int m0 = blockIdx.y * BM, n0 = blockIdx.x * BN;

  v8f acc[2][4] = {};

  // A-staging coords: 4096 elems / 256 thr = 16 contiguous halves each
  const int am = tid >> 1;
  const int ak = (tid & 1) * 16;
  // W-staging coords: 4096 elems / 256 thr = 16 each (one k-row, 16 n)
  const int wk  = tid >> 3;
  const int wnn = (tid & 7) * 16;

  for (int k0 = 0; k0 < K; k0 += BK) {
    // ---- stage A tile ----
    if constexpr (sizeof(TIn) == 4) {
      const float* ap = (const float*)A + (size_t)(m0 + am) * K + k0 + ak;
      union { unsigned short h[16]; uint4 q[2]; } pk;
      #pragma unroll
      for (int e = 0; e < 4; ++e) {
        float4 a4 = *(const float4*)(ap + e * 4);
        pk.h[e * 4 + 0] = f2bf(a4.x);
        pk.h[e * 4 + 1] = f2bf(a4.y);
        pk.h[e * 4 + 2] = f2bf(a4.z);
        pk.h[e * 4 + 3] = f2bf(a4.w);
      }
      *(uint4*)&As[am][ak]     = pk.q[0];
      *(uint4*)&As[am][ak + 8] = pk.q[1];
      if (k0 + BK < K) __builtin_prefetch(ap + BK, 0, 1);
    } else {
      const unsigned short* ap =
          (const unsigned short*)A + (size_t)(m0 + am) * K + k0 + ak;
      copy16_g2l(ap,     &As[am][ak]);
      copy16_g2l(ap + 8, &As[am][ak + 8]);
    }
    // ---- stage W tile transposed ----
    {
      const float* wp = W + (size_t)(k0 + wk) * N + n0 + wnn;
      #pragma unroll
      for (int e = 0; e < 4; ++e) {
        float4 w4 = *(const float4*)(wp + e * 4);
        Bt[wnn + e * 4 + 0][wk] = f2bf(w4.x);
        Bt[wnn + e * 4 + 1][wk] = f2bf(w4.y);
        Bt[wnn + e * 4 + 2][wk] = f2bf(w4.z);
        Bt[wnn + e * 4 + 3][wk] = f2bf(w4.w);
      }
      if (k0 + BK < K) __builtin_prefetch(wp + (size_t)BK * N, 0, 1);
    }
    if constexpr (sizeof(TIn) == 2) async_wait0();
    __syncthreads();

    // ---- fragments (all b128 LDS loads) ----
    FragAB af[2], bfr[4];
    #pragma unroll
    for (int i = 0; i < 2; ++i) {
      const int row = wm * 32 + i * 16 + ln;
      af[i].q[0] = *(const uint4*)&As[row][kh * 8];
      af[i].q[1] = *(const uint4*)&As[row][16 + kh * 8];
    }
    #pragma unroll
    for (int j = 0; j < 4; ++j) {
      const int col = wn * 64 + j * 16 + ln;
      bfr[j].q[0] = *(const uint4*)&Bt[col][kh * 8];
      bfr[j].q[1] = *(const uint4*)&Bt[col][16 + kh * 8];
    }
    #pragma unroll
    for (int i = 0; i < 2; ++i)
      #pragma unroll
      for (int j = 0; j < 4; ++j)
        acc[i][j] = wmma_bf16(af[i], bfr[j], acc[i][j]);
    __syncthreads();
  }

  // ---- epilogue: C layout M=(lane>>4)*8+r, N=lane&15 ----
  #pragma unroll
  for (int i = 0; i < 2; ++i) {
    #pragma unroll
    for (int j = 0; j < 4; ++j) {
      const int col = n0 + wn * 64 + j * 16 + ln;
      const float bv = bias[col];
      #pragma unroll
      for (int r = 0; r < 8; ++r) {
        const int row = m0 + wm * 32 + i * 16 + kh * 8 + r;
        const float val = (acc[i][j][r] + bv) * scale;
        if constexpr (OUT_BF16)
          ((unsigned short*)outp)[(size_t)row * N + col] = f2bf(val);
        else
          ((float*)outp)[(size_t)row * N + col] = val;
      }
    }
  }
}

// ---------------------------------------------------------------------------
// Causal flash attention over bf16 q/k/v planes laid out [B, S, H*64].
// Grid: (S/64, B*H); 128 threads = 4 waves, each wave owns 16 query rows.
// Per key-block of 64: QK^T = 8 WMMAs, PV = 8 WMMAs per wave.
// Q/K tiles staged via async-to-LDS (ASYNCcnt); V transposed manually.
// ---------------------------------------------------------------------------
__global__ __launch_bounds__(128) void flash_attn_kernel(
    const unsigned short* __restrict__ q, const unsigned short* __restrict__ k,
    const unsigned short* __restrict__ v, unsigned short* __restrict__ o) {
  __shared__ unsigned short Qs[64][64];      // 8 KB
  __shared__ unsigned short Ks[64][64];      // 8 KB  [key][dk]
  __shared__ unsigned short Vt[64][64];      // 8 KB  [dv][key] (transposed)
  __shared__ unsigned short Ps[4][16][64];   // 8 KB  per-wave P relayout

  const int tid = threadIdx.x;
  const int w = tid >> 5, lane = tid & 31;
  const int ln = lane & 15, kh = lane >> 4;
  const int b = blockIdx.y >> 4, h = blockIdx.y & 15;
  const int qblk = blockIdx.x;
  const int q0 = qblk * 64;
  const int SD = H_ * DK_;  // 1024

  // stage Q block (async b128 copies, already bf16)
  for (int f = tid * 8; f < 64 * 64; f += 128 * 8) {
    const int r = f >> 6, c = f & 63;
    copy16_g2l(&q[(size_t)(b * S_ + q0 + r) * SD + h * DK_ + c], &Qs[r][c]);
  }
  async_wait0();
  __syncthreads();

  FragAB aq[2];  // Q rows w*16..+15, K-dim split into two 32-chunks
  #pragma unroll
  for (int kb = 0; kb < 2; ++kb) {
    const int row = w * 16 + ln;
    aq[kb].q[0] = *(const uint4*)&Qs[row][kb * 32 + kh * 8];
    aq[kb].q[1] = *(const uint4*)&Qs[row][kb * 32 + 16 + kh * 8];
  }

  float mi[8], li[8];
  v8f oacc[4] = {};
  #pragma unroll
  for (int i = 0; i < 8; ++i) { mi[i] = -1e30f; li[i] = 0.f; }

  for (int j = 0; j <= qblk; ++j) {   // causal: only blocks at/below diagonal
    // stage K rows (async) + V transposed (manual)
    for (int f = tid * 8; f < 64 * 64; f += 128 * 8) {
      const int r = f >> 6, c = f & 63;
      const size_t base = (size_t)(b * S_ + j * 64 + r) * SD + h * DK_ + c;
      copy16_g2l(&k[base], &Ks[r][c]);
      union { uint4 u; unsigned short hh[8]; } tv;
      tv.u = *(const uint4*)&v[base];
      #pragma unroll
      for (int e = 0; e < 8; ++e) Vt[c + e][r] = tv.hh[e];
      if (j < qblk) {
        __builtin_prefetch(&k[base + (size_t)64 * SD], 0, 1);
        __builtin_prefetch(&v[base + (size_t)64 * SD], 0, 1);
      }
    }
    async_wait0();
    __syncthreads();

    // scores S = (Q/sqrt(d)) . K^T  (scale folded into q at projection)
    v8f s[4];
    #pragma unroll
    for (int nt = 0; nt < 4; ++nt) {
      v8f a = {};
      #pragma unroll
      for (int kb = 0; kb < 2; ++kb) {
        FragAB bk_;
        const int col = nt * 16 + ln;   // key index = WMMA N
        bk_.q[0] = *(const uint4*)&Ks[col][kb * 32 + kh * 8];
        bk_.q[1] = *(const uint4*)&Ks[col][kb * 32 + 16 + kh * 8];
        a = wmma_bf16(aq[kb], bk_, a);
      }
      s[nt] = a;
    }

    // causal mask inside diagonal block
    if (j == qblk) {
      #pragma unroll
      for (int nt = 0; nt < 4; ++nt)
        #pragma unroll
        for (int i = 0; i < 8; ++i) {
          const int mrow = w * 16 + kh * 8 + i;
          const int kcol = nt * 16 + ln;
          if (kcol > mrow) s[nt][i] = -1e30f;
        }
    }

    // online softmax; each row lives in 16 lanes (same kh group), vgpr i
    #pragma unroll
    for (int i = 0; i < 8; ++i) {
      float mx = fmaxf(fmaxf(s[0][i], s[1][i]), fmaxf(s[2][i], s[3][i]));
      #pragma unroll
      for (int d = 1; d < 16; d <<= 1) mx = fmaxf(mx, __shfl_xor(mx, d, 32));
      const float mnew = fmaxf(mi[i], mx);
      const float corr = __expf(mi[i] - mnew);
      float rs = 0.f;
      #pragma unroll
      for (int nt = 0; nt < 4; ++nt) {
        const float p = __expf(s[nt][i] - mnew);
        s[nt][i] = p;
        rs += p;
      }
      #pragma unroll
      for (int d = 1; d < 16; d <<= 1) rs += __shfl_xor(rs, d, 32);
      li[i] = li[i] * corr + rs;
      mi[i] = mnew;
      #pragma unroll
      for (int dt = 0; dt < 4; ++dt) oacc[dt][i] *= corr;
      // C-frag -> A-frag relayout via per-wave LDS scratch
      #pragma unroll
      for (int nt = 0; nt < 4; ++nt)
        Ps[w][kh * 8 + i][nt * 16 + ln] = f2bf(s[nt][i]);
    }

    asm volatile("s_wait_dscnt 0" ::: "memory");  // P visible to own wave

    // O += P . V
    #pragma unroll
    for (int kb = 0; kb < 2; ++kb) {
      FragAB ap;
      ap.q[0] = *(const uint4*)&Ps[w][ln][kb * 32 + kh * 8];
      ap.q[1] = *(const uint4*)&Ps[w][ln][kb * 32 + 16 + kh * 8];
      #pragma unroll
      for (int dt = 0; dt < 4; ++dt) {
        FragAB bv_;
        const int col = dt * 16 + ln;  // dv index = WMMA N
        bv_.q[0] = *(const uint4*)&Vt[col][kb * 32 + kh * 8];
        bv_.q[1] = *(const uint4*)&Vt[col][kb * 32 + 16 + kh * 8];
        oacc[dt] = wmma_bf16(ap, bv_, oacc[dt]);
      }
    }
    __syncthreads();
  }

  // normalize and write bf16 [B,S,H*64]
  #pragma unroll
  for (int i = 0; i < 8; ++i) {
    const float inv = 1.0f / li[i];
    const int row = q0 + w * 16 + kh * 8 + i;
    #pragma unroll
    for (int dt = 0; dt < 4; ++dt)
      o[(size_t)(b * S_ + row) * SD + h * DK_ + dt * 16 + ln] =
          f2bf(oacc[dt][i] * inv);
  }
}

// ---------------------------------------------------------------------------
extern "C" void kernel_launch(void* const* d_in, const int* in_sizes, int n_in,
                              void* d_out, int out_size, void* d_ws, size_t ws_size,
                              hipStream_t stream) {
  (void)in_sizes; (void)n_in; (void)out_size; (void)ws_size;
  const float* x  = (const float*)d_in[0];
  const float* Wq = (const float*)d_in[1];
  const float* bq = (const float*)d_in[2];
  const float* Wk = (const float*)d_in[3];
  const float* bk = (const float*)d_in[4];
  const float* Wv = (const float*)d_in[5];
  const float* bv = (const float*)d_in[6];
  const float* Wo = (const float*)d_in[7];
  const float* bo = (const float*)d_in[8];
  float* out = (float*)d_out;

  const int M = B_ * S_;                         // 4096
  const size_t plane = (size_t)M * D_;           // 4M elems (8 MB bf16 each)
  unsigned short* qb = (unsigned short*)d_ws;    // needs 32 MB workspace
  unsigned short* kb = qb + plane;
  unsigned short* vb = kb + plane;
  unsigned short* ab = vb + plane;

  dim3 gG(D_ / 128, M / 128);   // (8, 32)
  dim3 tG(256);

  // QKV projections (q pre-scaled by 1/sqrt(64))
  gemm_wmma_kernel<float, true><<<gG, tG, 0, stream>>>(x, Wq, bq, qb, M, D_, D_, 0.125f);
  gemm_wmma_kernel<float, true><<<gG, tG, 0, stream>>>(x, Wk, bk, kb, M, D_, D_, 1.0f);
  gemm_wmma_kernel<float, true><<<gG, tG, 0, stream>>>(x, Wv, bv, vb, M, D_, D_, 1.0f);

  // causal flash attention
  flash_attn_kernel<<<dim3(S_ / 64, B_ * H_), dim3(128), 0, stream>>>(qb, kb, vb, ab);

  // output projection -> fp32 d_out
  gemm_wmma_kernel<unsigned short, false><<<gG, tG, 0, stream>>>(ab, Wo, bo, out, M, D_, D_, 1.0f);
}
